// RN_38929583571428
// MI455X (gfx1250) — matrix operand order repaired and checked by reading, not compiled, not run on hardware
//
#include <hip/hip_runtime.h>

typedef _Float16 half_t;
typedef __attribute__((ext_vector_type(16))) _Float16 v16h;
typedef __attribute__((ext_vector_type(8)))  _Float16 v8h;
typedef __attribute__((ext_vector_type(8)))  float    v8f;
typedef __attribute__((ext_vector_type(4)))  float    v4f;
typedef __attribute__((ext_vector_type(4)))  unsigned int v4u;
typedef __attribute__((ext_vector_type(4)))  int      v4i;

typedef __attribute__((address_space(1))) v4i gv4i;   // global int4
typedef __attribute__((address_space(3))) v4i sv4i;   // LDS int4

#if defined(__has_builtin)
# if __has_builtin(__builtin_amdgcn_global_load_async_to_lds_b128) && \
     __has_builtin(__builtin_amdgcn_s_wait_asynccnt)
#  define USE_ASYNC_LDS 1
# endif
#endif

#define ROWH 264              // halves per LDS h-row (256 + 8 pad)
#define ROWB 528              // bytes per LDS h-row
#define HBUF_BYTES (256 * ROWB)          // 135168
#define WBUF_BYTES (256 * 256 * 2)       // 131072
#define FBUF_OFF   (HBUF_BYTES + WBUF_BYTES)
#define SMEM_TOTAL (FBUF_OFF + (256 + 256 + 32) * 4)

// ---------------- zero the scalar output ----------------
__global__ void k_zero(float* out) { out[0] = 0.0f; }

// ---------------- pack g_w2/3/4 into WMMA B layout, f16 ----------------
// Block (nt,kt) = 512 halves, split into two 256-half regions for conflict-free
// b128 LDS loads: region hg (0=lo 8 halves, 1=hi 8 halves), lane l at l*8.
// Lane l -> N = nt*16+(l&15), group g=l>>4; half h (0..15) -> K = kt*32 + h + 16*g.
__global__ void k_pack_w(const float* __restrict__ w2, const float* __restrict__ w3,
                         const float* __restrict__ w4, half_t* __restrict__ dst) {
    int gid = blockIdx.x * 256 + threadIdx.x;          // 3 * 65536 threads
    int w = gid >> 16;
    int e = gid & 65535;
    const float* src = (w == 0) ? w2 : (w == 1) ? w3 : w4;
    int block  = e >> 9;          // nt*8 + kt
    int within = e & 511;
    int hg  = within >> 8;        // half-group: 0 = halves 0..7, 1 = halves 8..15
    int rem = within & 255;
    int l   = rem >> 3;
    int h   = hg * 8 + (rem & 7);
    int nt = block >> 3;
    int kt = block & 7;
    int K = kt * 32 + h + 16 * (l >> 4);
    int N = nt * 16 + (l & 15);
    dst[gid] = (half_t)src[K * 256 + N];
}

// ---------------- precompute A = sup_feat @ W1[:258], B = qry_feat @ W1[258:] ----------------
__global__ void k_build_ab(const float* __restrict__ sup, const float* __restrict__ qry,
                           const float* __restrict__ gw1,
                           float* __restrict__ Ag, float* __restrict__ Bg) {
    int bid = blockIdx.x;        // 640 A-rows then 1920 B-rows
    int o = threadIdx.x;
    if (bid < 640) {
        int bs = bid >> 4, n = bid & 15;
        const float* base = sup + (size_t)bs * 256 * 16;
        float acc = 0.0f;
        for (int ch = 0; ch < 256; ++ch)
            acc += base[ch * 16 + n] * gw1[ch * 256 + o];
        acc += ((float)(n >> 2) * 0.25f) * gw1[256 * 256 + o];
        acc += ((float)(n & 3) * 0.25f)  * gw1[257 * 256 + o];
        Ag[(size_t)bid * 256 + o] = acc;
    } else {
        int r = bid - 640;
        int bq = r >> 4, m = r & 15;
        const float* base = qry + (size_t)bq * 256 * 16;
        float acc = 0.0f;
        for (int ch = 0; ch < 256; ++ch)
            acc += base[ch * 16 + m] * gw1[(258 + ch) * 256 + o];
        acc += ((float)(m >> 2) * 0.25f) * gw1[514 * 256 + o];
        acc += ((float)(m & 3) * 0.25f)  * gw1[515 * 256 + o];
        Bg[(size_t)r * 256 + o] = acc;
    }
}

// ---------------- fused main kernel: one (b,q,s) triple per workgroup ----------------
__global__ __launch_bounds__(256)
void k_main(const float* __restrict__ Ag, const float* __restrict__ Bg,
            const float* __restrict__ gb1, const half_t* __restrict__ packW,
            const float* __restrict__ gb2, const float* __restrict__ gb3,
            const float* __restrict__ gb4,
            const float* __restrict__ fw1, const float* __restrict__ fb1,
            const float* __restrict__ fw2, const float* __restrict__ fb2,
            const float* __restrict__ fw3, const float* __restrict__ fb3,
            const float* __restrict__ fw4, const float* __restrict__ fb4,
            const int* __restrict__ sup_y, const int* __restrict__ qry_y,
            float* __restrict__ out) {
    extern __shared__ __align__(16) char smem[];
    half_t* hbuf = (half_t*)smem;
    half_t* wbuf = (half_t*)(smem + HBUF_BYTES);
    float*  partf = (float*)(smem + HBUF_BYTES);    // reused after GEMMs (8x256 f32)
    float*  fa   = (float*)(smem + FBUF_OFF);
    float*  fbv  = fa + 256;
    float*  y3b  = fbv + 256;

    const int bid = blockIdx.x;          // 0..599
    const int s_i = bid % 5;
    const int q_i = (bid / 5) % 15;
    const int b_i = bid / 75;

    const int tid = threadIdx.x;
    const int wv = tid >> 5;
    const int ln = tid & 31;
    const int lu = ln & 15;
    const int g  = ln >> 4;

    // ---- Phase A: h1[m*16+n, k] = relu(A[s,n,k] + B[q,m,k] + b1[k]) ----
    {
        const float* Arow = Ag + ((size_t)(b_i * 5 + s_i) * 16) * 256;
        const float* Brow = Bg + ((size_t)(b_i * 15 + q_i) * 16) * 256;
        v4f g0 = *(const v4f*)(gb1 + ln * 8);
        v4f g1 = *(const v4f*)(gb1 + ln * 8 + 4);
        for (int r = wv * 32; r < wv * 32 + 32; ++r) {
            int m = r >> 4, n = r & 15;
            const v4f* Ap = (const v4f*)(Arow + n * 256 + ln * 8);
            const v4f* Bp = (const v4f*)(Brow + m * 256 + ln * 8);
            v4f a0 = Ap[0], a1 = Ap[1], b0 = Bp[0], b1 = Bp[1];
            v8h hv;
#pragma unroll
            for (int i = 0; i < 4; ++i) {
                float x0 = a0[i] + b0[i] + g0[i];
                float x1 = a1[i] + b1[i] + g1[i];
                hv[i]     = (half_t)(x0 > 0.0f ? x0 : 0.0f);
                hv[4 + i] = (half_t)(x1 > 0.0f ? x1 : 0.0f);
            }
            *(v8h*)(hbuf + (size_t)r * ROWH + ln * 8) = hv;
        }
    }

    // ---- Phase B: three 256x256x256 GEMM layers via WMMA, in-place per wave slab ----
    for (int layer = 0; layer < 3; ++layer) {
        __syncthreads();   // all waves done reading previous wbuf
#ifdef USE_ASYNC_LDS
        {                  // async DMA: global packed weights -> LDS (ASYNCcnt path)
            gv4i* gsrc = (gv4i*)(packW + (size_t)layer * 65536);
            sv4i* sdst = (sv4i*)wbuf;
            for (int i = tid; i < WBUF_BYTES / 16; i += 256)
                __builtin_amdgcn_global_load_async_to_lds_b128(gsrc + i, sdst + i, 0, 0);
            __builtin_amdgcn_s_wait_asynccnt(0);
        }
#else
        {
            const v4u* src = (const v4u*)(packW + (size_t)layer * 65536);
            v4u* dst = (v4u*)wbuf;
            for (int i = tid; i < WBUF_BYTES / 16; i += 256) dst[i] = src[i];
        }
#endif
        if (layer < 2)     // pull next layer's weights toward L2/L0
            __builtin_prefetch((const char*)packW + (size_t)(layer + 1) * 131072 + tid * 512, 0, 1);
        __syncthreads();
        const float* gb = (layer == 0) ? gb2 : (layer == 1) ? gb3 : gb4;

        for (int mi = 0; mi < 2; ++mi) {
            const int m0 = (wv * 2 + mi) * 16;
            // load A tiles for this 16-row block (full K=256) into registers
            v16h a[8];
            const char* arow = (const char*)hbuf + (size_t)(m0 + lu) * ROWB + g * 32;
#pragma unroll
            for (int kt = 0; kt < 8; ++kt) {
                ((v8h*)&a[kt])[0] = *(const v8h*)(arow + kt * 64);
                ((v8h*)&a[kt])[1] = *(const v8h*)(arow + kt * 64 + 16);
            }
            // two independent accumulator chains per iteration (better XDL overlap)
            for (int nt = 0; nt < 16; nt += 2) {
                v8f c0 = {}, c1 = {};
#pragma unroll
                for (int kt = 0; kt < 8; ++kt) {
                    const half_t* wb0 = wbuf + (size_t)(nt * 8 + kt) * 512;
                    v16h bm0, bm1;
                    ((v8h*)&bm0)[0] = *(const v8h*)(wb0 + ln * 8);
                    ((v8h*)&bm0)[1] = *(const v8h*)(wb0 + 256 + ln * 8);
                    ((v8h*)&bm1)[0] = *(const v8h*)(wb0 + 4096 + ln * 8);
                    ((v8h*)&bm1)[1] = *(const v8h*)(wb0 + 4096 + 256 + ln * 8);
                    c0 = __builtin_amdgcn_wmma_f32_16x16x32_f16(
                        false, a[kt], false, bm0, (short)0, c0, false, false);
                    c1 = __builtin_amdgcn_wmma_f32_16x16x32_f16(
                        false, a[kt], false, bm1, (short)0, c1, false, false);
                }
                float bv0 = gb[nt * 16 + lu];
                float bv1 = gb[nt * 16 + 16 + lu];
                // base + constant row-step stores: fold v*ROWB into DS imm offsets
                char* srow = (char*)hbuf + (size_t)(m0 + 8 * g) * ROWB + lu * 2 + nt * 32;
#pragma unroll
                for (int v = 0; v < 8; ++v) {
                    float x0 = c0[v] + bv0; x0 = x0 > 0.0f ? x0 : 0.0f;
                    float x1 = c1[v] + bv1; x1 = x1 > 0.0f ? x1 : 0.0f;
                    *(half_t*)(srow + v * ROWB)      = (half_t)x0;
                    *(half_t*)(srow + v * ROWB + 32) = (half_t)x1;
                }
            }
        }
    }
    __syncthreads();       // GEMMs done; wbuf region free for reuse

    // ---- Phase C: sum over 256 pair rows (wave-parallel, b128 row reads) ----
    {
        float acc[8];
#pragma unroll
        for (int i = 0; i < 8; ++i) acc[i] = 0.0f;
        for (int r = wv * 32; r < wv * 32 + 32; ++r) {
            v8h hv = *(const v8h*)(hbuf + (size_t)r * ROWH + ln * 8);
#pragma unroll
            for (int i = 0; i < 8; ++i) acc[i] += (float)hv[i];
        }
#pragma unroll
        for (int i = 0; i < 8; ++i) partf[wv * 256 + ln * 8 + i] = acc[i];
    }
    __syncthreads();
    {
        float acc = 0.0f;
#pragma unroll
        for (int w = 0; w < 8; ++w) acc += partf[w * 256 + tid];
        fa[tid] = acc;
    }
    __syncthreads();
    {   // y1 = relu(xf @ fw1 + fb1)
        float acc = fb1[tid];
        for (int k = 0; k < 256; ++k) acc += fa[k] * fw1[k * 256 + tid];
        fbv[tid] = acc > 0.0f ? acc : 0.0f;
    }
    __syncthreads();
    {   // y2 = relu(y1 @ fw2 + fb2)
        float acc = fb2[tid];
        for (int k = 0; k < 256; ++k) acc += fbv[k] * fw2[k * 256 + tid];
        fa[tid] = acc > 0.0f ? acc : 0.0f;
    }
    __syncthreads();
    if (tid < 29) {   // y3 = relu(y2 @ fw3 + fb3)
        float acc = fb3[tid];
        for (int k = 0; k < 256; ++k) acc += fa[k] * fw3[k * 29 + tid];
        y3b[tid] = acc > 0.0f ? acc : 0.0f;
    }
    __syncthreads();
    if (tid == 0) {
        float sc = fb4[0];
        for (int k = 0; k < 29; ++k) sc += y3b[k] * fw4[k];
        float score = 1.0f / (1.0f + __expf(-sc));
        float label = (sup_y[b_i * 5 + s_i] == qry_y[b_i * 15 + q_i]) ? 1.0f : 0.0f;
        float d = label - score;
        atomicAdd(out, d * d * 0.125f);   // / b
    }
}

extern "C" void kernel_launch(void* const* d_in, const int* in_sizes, int n_in,
                              void* d_out, int out_size, void* d_ws, size_t ws_size,
                              hipStream_t stream) {
    const float* support_xf = (const float*)d_in[0];
    const float* query_xf   = (const float*)d_in[1];
    const int*   sup_y      = (const int*)d_in[2];
    const int*   qry_y      = (const int*)d_in[3];
    const float* gw1 = (const float*)d_in[4];
    const float* gb1 = (const float*)d_in[5];
    const float* gw2 = (const float*)d_in[6];
    const float* gb2 = (const float*)d_in[7];
    const float* gw3 = (const float*)d_in[8];
    const float* gb3 = (const float*)d_in[9];
    const float* gw4 = (const float*)d_in[10];
    const float* gb4 = (const float*)d_in[11];
    const float* fw1 = (const float*)d_in[12];
    const float* fb1 = (const float*)d_in[13];
    const float* fw2 = (const float*)d_in[14];
    const float* fb2 = (const float*)d_in[15];
    const float* fw3 = (const float*)d_in[16];
    const float* fb3 = (const float*)d_in[17];
    const float* fw4 = (const float*)d_in[18];
    const float* fb4 = (const float*)d_in[19];

    // workspace: [packed weights f16: 3*65536*2 B][Ag: 640*256*4 B][Bg: 1920*256*4 B]
    half_t* packW = (half_t*)d_ws;
    float*  Ag    = (float*)((char*)d_ws + 393216);
    float*  Bg    = (float*)((char*)d_ws + 1048576);

    k_zero<<<1, 1, 0, stream>>>((float*)d_out);
    k_pack_w<<<768, 256, 0, stream>>>(gw2, gw3, gw4, packW);
    k_build_ab<<<2560, 256, 0, stream>>>(support_xf, query_xf, gw1, Ag, Bg);
    k_main<<<600, 256, SMEM_TOTAL, stream>>>(
        Ag, Bg, gb1, packW, gb2, gb3, gb4,
        fw1, fb1, fw2, fb2, fw3, fb3, fw4, fb4,
        sup_y, qry_y, (float*)d_out);
}
